// MobileViTv2Attention_82703890252111
// MI455X (gfx1250) — matrix operand order repaired and checked by reading, not compiled, not run on hardware
//
#include <hip/hip_runtime.h>
#include <hip/hip_bf16.h>

#define DD    1024
#define NSEQ  4096
#define BATCH 8
#define ROWS  (BATCH * NSEQ)
#define ROWB  40                      // LDS B-panel row stride in bf16 (80 bytes, padded)
#define PANEL (64 * ROWB)             // one K-step B panel in LDS (elements)
#define NK    (DD / 32)               // K-steps

typedef __bf16 bf16_t;
typedef __attribute__((ext_vector_type(16))) __bf16 v16bf;
typedef __attribute__((ext_vector_type(8)))  __bf16 v8bf;
typedef __attribute__((ext_vector_type(8)))  float  v8f;
typedef __attribute__((ext_vector_type(4)))  float  v4f;

#if defined(__has_builtin)
#if __has_builtin(__builtin_amdgcn_tensor_load_to_lds) && __has_builtin(__builtin_amdgcn_s_wait_tensorcnt)
#define USE_TDM 1
#endif
#endif

#ifdef USE_TDM
typedef unsigned int u32x4 __attribute__((ext_vector_type(4)));
typedef int          i32x4 __attribute__((ext_vector_type(4)));
typedef int          i32x8 __attribute__((ext_vector_type(8)));

// TDM: DMA a 64(row) x 32(k) bf16 tile from a [1024 x 1024] bf16 tensor into LDS,
// with hardware padding: 4 DWORDs inserted after every 16 DWORDs (one 64B row)
// -> LDS row stride 80B (20 banks; conflict-free for 16 concurrent columns).
__device__ inline void tdm_load_panel(const bf16_t* gsrc, unsigned lds_byte_addr) {
    unsigned long long ga = (unsigned long long)(uintptr_t)gsrc;
    u32x4 g0;
    g0[0] = 1u;                                            // count=1, user mode
    g0[1] = lds_byte_addr;                                 // lds_addr
    g0[2] = (unsigned)ga;                                  // global_addr[31:0]
    g0[3] = ((unsigned)(ga >> 32) & 0x1FFFFFFu) | (2u << 30); // addr[56:32] | type=2
    i32x8 g1;
    g1[0] = (1 << 16) | (1 << 20) | (3 << 22) | (3 << 25); // data_size=2B, pad_en, ivl=16DW, amt=4DW
    g1[1] = (int)(1024u << 16);                            // tensor_dim0 = 1024 (bits 79:48)
    g1[2] = (int)(1024u << 16);                            // tensor_dim1 = 1024 (bits 111:80)
    g1[3] = (int)(32u << 16);                              // tile_dim0 = 32 (bits 127:112)
    g1[4] = 64;                                            // tile_dim1 = 64, tile_dim2 = 0
    g1[5] = 1024;                                          // tensor_dim0_stride = 1024
    g1[6] = 0;
    g1[7] = 0;
    i32x4 z4 = {0, 0, 0, 0};
#if __clang_major__ >= 23
    i32x8 z8 = {0, 0, 0, 0, 0, 0, 0, 0};
    __builtin_amdgcn_tensor_load_to_lds(g0, g1, z4, z4, z8, 0);
#else
    __builtin_amdgcn_tensor_load_to_lds(g0, g1, z4, z4, 0);
#endif
}
#endif

// Cooperative fallback: 256 threads stage the 64x32 panel (same padded layout).
__device__ inline void coop_load_panel(const bf16_t* __restrict__ gsrc, bf16_t* lbuf) {
    int t = threadIdx.x;
    int c = t >> 2, q = t & 3;
    v8bf d = *(const v8bf*)(gsrc + (size_t)c * DD + q * 8);
    *(v8bf*)(lbuf + c * ROWB + q * 8) = d;
}

// ---------------- logits: i[b,n] = dot(x[b,n,:], Wi) + bi  (wave per token) ----
__global__ void k_logits(const float* __restrict__ x, const float* __restrict__ Wi,
                         const float* __restrict__ bi, float* __restrict__ logits) {
    int wave = (blockIdx.x * blockDim.x + threadIdx.x) >> 5;
    int lane = threadIdx.x & 31;
    if (wave >= ROWS) return;
    const float* xr = x + (size_t)wave * DD;
    float s = 0.f;
    #pragma unroll 4
    for (int j = lane; j < DD; j += 32) s += xr[j] * Wi[j];
    #pragma unroll
    for (int off = 16; off > 0; off >>= 1) s += __shfl_xor(s, off, 32);
    if (lane == 0) logits[wave] = s + bi[0];
}

// ---------------- softmax over sequence, one block per batch ------------------
__global__ void k_softmax(const float* __restrict__ logits, float* __restrict__ w) {
    __shared__ float red[256];
    int b = blockIdx.x, t = threadIdx.x;
    const float* l = logits + b * NSEQ;
    float m = -3.402823466e38f;
    for (int n = t; n < NSEQ; n += 256) m = fmaxf(m, l[n]);
    red[t] = m; __syncthreads();
    for (int s = 128; s > 0; s >>= 1) { if (t < s) red[t] = fmaxf(red[t], red[t + s]); __syncthreads(); }
    m = red[0]; __syncthreads();
    float sum = 0.f;
    for (int n = t; n < NSEQ; n += 256) sum += __expf(l[n] - m);
    red[t] = sum; __syncthreads();
    for (int s = 128; s > 0; s >>= 1) { if (t < s) red[t] += red[t + s]; __syncthreads(); }
    float inv = 1.f / red[0];
    for (int n = t; n < NSEQ; n += 256) w[b * NSEQ + n] = __expf(l[n] - m) * inv;
}

// ---------------- xbar[b,d] = sum_n w[b,n] * x[b,n,d] -------------------------
__global__ void k_xbar(const float* __restrict__ x, const float* __restrict__ w,
                       float* __restrict__ xbar) {
    int b = blockIdx.y;
    int d = blockIdx.x * 256 + threadIdx.x;
    const float* xb = x + (size_t)b * NSEQ * DD;
    const float* wb = w + b * NSEQ;
    float acc = 0.f;
    for (int n = 0; n < NSEQ; ++n) acc += wb[n] * xb[(size_t)n * DD + d];
    xbar[b * DD + d] = acc;
}

// ---------------- context[b,e] = xbar[b,:] . Wk[:,e] + bk[e] ------------------
__global__ void k_context(const float* __restrict__ xbar, const float* __restrict__ Wk,
                          const float* __restrict__ bk, float* __restrict__ ctx) {
    int b = blockIdx.y;
    int e = blockIdx.x * 256 + threadIdx.x;
    const float* xb = xbar + b * DD;
    float acc = bk[e];
    for (int d = 0; d < DD; ++d) acc += xb[d] * Wk[(size_t)d * DD + e];
    ctx[b * DD + e] = acc;
}

// ---------------- transpose + fp32 -> bf16 convert of a DxD weight ------------
__global__ void k_transpose_cvt(const float* __restrict__ W, bf16_t* __restrict__ Wt) {
    __shared__ float tile[32][33];
    int k0 = blockIdx.x * 32, n0 = blockIdx.y * 32;
    int tx = threadIdx.x & 31, ty = threadIdx.x >> 5;   // 32 x 8
    #pragma unroll
    for (int r = 0; r < 32; r += 8)
        tile[ty + r][tx] = W[(size_t)(k0 + ty + r) * DD + n0 + tx];
    __syncthreads();
    #pragma unroll
    for (int r = 0; r < 32; r += 8)
        Wt[(size_t)(n0 + ty + r) * DD + k0 + tx] = (bf16_t)tile[tx][ty + r];
}

// B operand (32x16) read from padded LDS panel: lane col = l15, K = half*8..+7, 16+half*8..+7
__device__ inline v16bf lds_read_b(const bf16_t* cur, int jcol, int l15, int half) {
    const bf16_t* bp = cur + (size_t)(jcol * 16 + l15) * ROWB + half * 8;
    v8bf b0 = *(const v8bf*)bp;
    v8bf b1 = *(const v8bf*)(bp + 16);
    v16bf bm;
    #pragma unroll
    for (int e = 0; e < 8; ++e) { bm[e] = b0[e]; bm[8 + e] = b1[e]; }
    return bm;
}

// ---------------- GEMM1: v = bf16((x @ Wv + bv) * ctx) ------------------------
// Workgroup: 8 waves = 128x64 tile. B panel staged in LDS by TDM, double-buffered.
// A layout (16x32 bf16, ISA 7.12.2): lane = half*16+m; elems 0..7 -> K=half*8..,
// elems 8..15 -> K=16+half*8..
__global__ void __launch_bounds__(256)
k_gemm_v(const float* __restrict__ x, const bf16_t* __restrict__ Wvt,
         const float* __restrict__ bv, const float* __restrict__ ctx,
         bf16_t* __restrict__ vout) {
    __shared__ bf16_t smem[2 * PANEL];
    int w = threadIdx.x >> 5, lane = threadIdx.x & 31;
    int row16 = blockIdx.y * 128 + w * 16;      // per-wave 16 rows
    int n0 = blockIdx.x * 64;                   // col tile (fast grid dim -> A rows L2-hot)
    int half = lane >> 4, l15 = lane & 15;
    bool lead = (threadIdx.x < 32);
    const bf16_t* Bsrc = Wvt + (size_t)n0 * DD; // row-major [n][k], tile = 64 rows x 32 k
    const float* arow = x + (size_t)(row16 + l15) * DD;

    v8f acc[4] = {};
#ifdef USE_TDM
    if (lead) tdm_load_panel(Bsrc, (unsigned)(uintptr_t)&smem[0]);
#endif
    for (int it = 0; it < NK; ++it) {
        int kt = it * 32;
        bf16_t* cur = smem + (it & 1) * PANEL;
#ifdef USE_TDM
        if (it + 1 < NK) {
            if (lead) {
                tdm_load_panel(Bsrc + (kt + 32), (unsigned)(uintptr_t)&smem[((it + 1) & 1) * PANEL]);
                __builtin_amdgcn_s_wait_tensorcnt(1);   // current panel landed
            }
        } else if (lead) {
            __builtin_amdgcn_s_wait_tensorcnt(0);
        }
        __syncthreads();
#else
        __syncthreads();
        cur = smem;
        coop_load_panel(Bsrc + kt, cur);
        __syncthreads();
#endif
        // A: 16 fp32 -> bf16 per lane
        v4f a0 = *(const v4f*)(arow + kt + half * 8);
        v4f a1 = *(const v4f*)(arow + kt + half * 8 + 4);
        v4f a2 = *(const v4f*)(arow + kt + 16 + half * 8);
        v4f a3 = *(const v4f*)(arow + kt + 16 + half * 8 + 4);
        v16bf a;
        #pragma unroll
        for (int j = 0; j < 4; ++j) {
            a[j]      = (bf16_t)a0[j];
            a[4 + j]  = (bf16_t)a1[j];
            a[8 + j]  = (bf16_t)a2[j];
            a[12 + j] = (bf16_t)a3[j];
        }
        #pragma unroll
        for (int j = 0; j < 4; ++j) {
            v16bf bm = lds_read_b(cur, j, l15, half);
            acc[j] = __builtin_amdgcn_wmma_f32_16x16x32_bf16(false, a, false, bm,
                                                             (short)0, acc[j], false, false);
        }
#ifdef USE_TDM
        __syncthreads();   // reads of cur done before it is overwritten next+1 step
#endif
    }
    int rbase = row16 + half * 8;
    int b = row16 >> 12;                        // 4096 rows per batch
    #pragma unroll
    for (int j = 0; j < 4; ++j) {
        int col = n0 + j * 16 + l15;
        float bvc = bv[col];
        float cc  = ctx[b * DD + col];
        #pragma unroll
        for (int r = 0; r < 8; ++r)
            vout[(size_t)(rbase + r) * DD + col] = (bf16_t)((acc[j][r] + bvc) * cc);
    }
}

// ---------------- GEMM2: out = v @ Wo + bo (fp32 out) -------------------------
__global__ void __launch_bounds__(256)
k_gemm_o(const bf16_t* __restrict__ vin, const bf16_t* __restrict__ Wot,
         const float* __restrict__ bo, float* __restrict__ out) {
    __shared__ bf16_t smem[2 * PANEL];
    int w = threadIdx.x >> 5, lane = threadIdx.x & 31;
    int row16 = blockIdx.y * 128 + w * 16;
    int n0 = blockIdx.x * 64;
    int half = lane >> 4, l15 = lane & 15;
    bool lead = (threadIdx.x < 32);
    const bf16_t* Bsrc = Wot + (size_t)n0 * DD;
    const bf16_t* arow = vin + (size_t)(row16 + l15) * DD;

    v8f acc[4] = {};
#ifdef USE_TDM
    if (lead) tdm_load_panel(Bsrc, (unsigned)(uintptr_t)&smem[0]);
#endif
    for (int it = 0; it < NK; ++it) {
        int kt = it * 32;
        bf16_t* cur = smem + (it & 1) * PANEL;
#ifdef USE_TDM
        if (it + 1 < NK) {
            if (lead) {
                tdm_load_panel(Bsrc + (kt + 32), (unsigned)(uintptr_t)&smem[((it + 1) & 1) * PANEL]);
                __builtin_amdgcn_s_wait_tensorcnt(1);
            }
        } else if (lead) {
            __builtin_amdgcn_s_wait_tensorcnt(0);
        }
        __syncthreads();
#else
        __syncthreads();
        cur = smem;
        coop_load_panel(Bsrc + kt, cur);
        __syncthreads();
#endif
        v8bf a0 = *(const v8bf*)(arow + kt + half * 8);
        v8bf a1 = *(const v8bf*)(arow + kt + 16 + half * 8);
        v16bf a;
        #pragma unroll
        for (int j = 0; j < 8; ++j) { a[j] = a0[j]; a[8 + j] = a1[j]; }
        #pragma unroll
        for (int j = 0; j < 4; ++j) {
            v16bf bm = lds_read_b(cur, j, l15, half);
            acc[j] = __builtin_amdgcn_wmma_f32_16x16x32_bf16(false, a, false, bm,
                                                             (short)0, acc[j], false, false);
        }
#ifdef USE_TDM
        __syncthreads();
#endif
    }
    int rbase = row16 + half * 8;
    #pragma unroll
    for (int j = 0; j < 4; ++j) {
        int col = n0 + j * 16 + l15;
        float boc = bo[col];
        #pragma unroll
        for (int r = 0; r < 8; ++r)
            out[(size_t)(rbase + r) * DD + col] = acc[j][r] + boc;
    }
}

extern "C" void kernel_launch(void* const* d_in, const int* in_sizes, int n_in,
                              void* d_out, int out_size, void* d_ws, size_t ws_size,
                              hipStream_t stream) {
    const float* x  = (const float*)d_in[0];
    const float* Wi = (const float*)d_in[1];
    const float* bi = (const float*)d_in[2];
    const float* Wk = (const float*)d_in[3];
    const float* bk = (const float*)d_in[4];
    const float* Wv = (const float*)d_in[5];
    const float* bv = (const float*)d_in[6];
    const float* Wo = (const float*)d_in[7];
    const float* bo = (const float*)d_in[8];
    float* out = (float*)d_out;

    char* w8 = (char*)d_ws;
    float*  logits = (float*)(w8);                              // 128 KB
    float*  wsm    = (float*)(w8 + 131072);                     // 128 KB
    float*  xbar   = (float*)(w8 + 262144);                     // 32 KB
    float*  ctx    = (float*)(w8 + 294912);                     // 32 KB
    bf16_t* Wvt    = (bf16_t*)(w8 + 327680);                    // 2 MB
    bf16_t* Wot    = (bf16_t*)(w8 + 327680 + 2097152);          // 2 MB
    bf16_t* vbuf   = (bf16_t*)(w8 + 327680 + 2 * 2097152);      // 64 MB

    // 1) logits (wave per token)
    k_logits<<<ROWS / 8, 256, 0, stream>>>(x, Wi, bi, logits);
    // 2) softmax per batch
    k_softmax<<<BATCH, 256, 0, stream>>>(logits, wsm);
    // 3) weighted row-sum  xbar = w^T x
    k_xbar<<<dim3(DD / 256, BATCH), 256, 0, stream>>>(x, wsm, xbar);
    // 4) context = xbar @ Wk + bk  (folded k-GEMM)
    k_context<<<dim3(DD / 256, BATCH), 256, 0, stream>>>(xbar, Wk, bk, ctx);
    // 5) weight transpose + bf16 convert
    k_transpose_cvt<<<dim3(DD / 32, DD / 32), 256, 0, stream>>>(Wv, Wvt);
    k_transpose_cvt<<<dim3(DD / 32, DD / 32), 256, 0, stream>>>(Wo, Wot);
    // 6/7) the two big GEMMs: 128x64 tile per workgroup, TDM-staged B panels
    dim3 gg(DD / 64, ROWS / 128);   // (16, 256)
    k_gemm_v<<<gg, 256, 0, stream>>>(x, Wvt, bv, ctx, vbuf);
    k_gemm_o<<<gg, 256, 0, stream>>>(vbuf, Wot, bo, out);
}
